// EquivariantMultiHeadAttention_83004537963112
// MI455X (gfx1250) — compile-verified, matrix operand order checked by reading, not run on hardware
//
#include <hip/hip_runtime.h>
#include <cstddef>
#include <cstdint>

// Problem constants (match reference setup_inputs)
#define N_NODES 32768
#define N_EDGES 262144
#define HC      128
#define RBFDIM  64
#define ECHUNK  32768
#define NCHUNK  (N_EDGES / ECHUNK)

typedef __attribute__((ext_vector_type(16))) __bf16 v16bf;
typedef __attribute__((ext_vector_type(8)))  __bf16 v8bf;
typedef __attribute__((ext_vector_type(8)))  float  v8f;

// ---------------------------------------------------------------------------
// WMMA fragment loader: 16x32 bf16 tile in A-layout.
// lane m in [0,16): row = rowBase+m, halves 0..7 = K kBase+hi*8..+7,
// halves 8..15 = K kBase+16+hi*8..+7  (hi = lane>>4). Two b128 loads.
// ---------------------------------------------------------------------------
static __device__ __forceinline__ v16bf tmd_load_frag(const __bf16* base, int ld,
                                                      int rowBase, int kBase, int lane) {
  int m  = lane & 15;
  int hi = lane >> 4;
  const __bf16* p = base + (size_t)(rowBase + m) * (size_t)ld + kBase + hi * 8;
  v8bf lo = *(const v8bf*)(p);
  v8bf hh = *(const v8bf*)(p + 16);
  return __builtin_shufflevector(lo, hh, 0,1,2,3,4,5,6,7,8,9,10,11,12,13,14,15);
}

// ---------------------------------------------------------------------------
// Y[r][c] = act( sum_k X[r][k] * W[c][k] + bias[c] )
// X:[R,K] bf16 row-major, W:[C,K] bf16 row-major, Y:[R,C] f32.
// grid = (C/16, R/128), block = 256 (8 waves; wave w -> 16-row slice).
// act: 0 = identity, 1 = SiLU.
// ---------------------------------------------------------------------------
__global__ __launch_bounds__(256)
void tmd_gemm_bf16_nt(const __bf16* __restrict__ X, const __bf16* __restrict__ W,
                      const float* __restrict__ bias, float* __restrict__ Y,
                      int K, int C, int act) {
  int lane = threadIdx.x & 31;
  int wave = threadIdx.x >> 5;
  int rowBase = (blockIdx.y * 8 + wave) * 16;
  int colBase = blockIdx.x * 16;

  v8f acc = {};
  for (int kb = 0; kb < K; kb += 32) {
    v16bf a = tmd_load_frag(X, K, rowBase, kb, lane);
    v16bf b = tmd_load_frag(W, K, colBase, kb, lane);
    acc = __builtin_amdgcn_wmma_f32_16x16x32_bf16(
        /*neg_a=*/false, a, /*neg_b=*/false, b,
        /*c_mod=*/(short)0, acc, /*reuse_a=*/false, /*reuse_b=*/false);
  }

  int n  = lane & 15;
  int hi = lane >> 4;
  float bv = bias ? bias[colBase + n] : 0.0f;
#pragma unroll
  for (int r = 0; r < 8; ++r) {
    float y = acc[r] + bv;
    if (act) y = y / (1.0f + __expf(-y));      // SiLU
    Y[(size_t)(rowBase + r + 8 * hi) * (size_t)C + colBase + n] = y;
  }
}

// ---------------------------------------------------------------------------
// LayerNorm over HC=128, one wave per row, output bf16.
// ---------------------------------------------------------------------------
__global__ __launch_bounds__(256)
void tmd_ln_to_bf16(const float* __restrict__ x, const float* __restrict__ g,
                    const float* __restrict__ b, __bf16* __restrict__ xn, int nrows) {
  int gid  = blockIdx.x * blockDim.x + threadIdx.x;
  int row  = gid >> 5;
  int lane = gid & 31;
  if (row >= nrows) return;
  float4 v = *(const float4*)(x + (size_t)row * HC + lane * 4);
  float s = v.x + v.y + v.z + v.w;
#pragma unroll
  for (int o = 16; o > 0; o >>= 1) s += __shfl_xor(s, o, 32);
  float mu = s * (1.0f / 128.0f);
  float d0 = v.x - mu, d1 = v.y - mu, d2 = v.z - mu, d3 = v.w - mu;
  float vs = d0 * d0 + d1 * d1 + d2 * d2 + d3 * d3;
#pragma unroll
  for (int o = 16; o > 0; o >>= 1) vs += __shfl_xor(vs, o, 32);
  float inv = rsqrtf(vs * (1.0f / 128.0f) + 1e-5f);
  float4 gg = *(const float4*)(g + lane * 4);
  float4 bb = *(const float4*)(b + lane * 4);
  __bf16* o4 = xn + (size_t)row * HC + lane * 4;
  o4[0] = (__bf16)(d0 * inv * gg.x + bb.x);
  o4[1] = (__bf16)(d1 * inv * gg.y + bb.y);
  o4[2] = (__bf16)(d2 * inv * gg.z + bb.z);
  o4[3] = (__bf16)(d3 * inv * gg.w + bb.w);
}

// ---------------------------------------------------------------------------
// Elementwise helpers
// ---------------------------------------------------------------------------
__global__ void tmd_cvt_bf16(const float* __restrict__ s, __bf16* __restrict__ d, size_t n) {
  size_t i = (size_t)blockIdx.x * blockDim.x + threadIdx.x;
  if (i < n) d[i] = (__bf16)s[i];
}

__global__ void tmd_zero_f32(float* __restrict__ p, size_t n) {
  size_t i = (size_t)blockIdx.x * blockDim.x + threadIdx.x;
  if (i < n) p[i] = 0.0f;
}

static __device__ __forceinline__ void tmd_afa(float* p, float v) {
  __hip_atomic_fetch_add(p, v, __ATOMIC_RELAXED, __HIP_MEMORY_SCOPE_AGENT);
}

// ---------------------------------------------------------------------------
// Edge message + scatter. One wave32 per edge; lane owns 4 contiguous channels.
// qkv layout per node: [0,128)=q, [128,256)=k, [256,640)=v.
// dkv layout per edge (chunk-local): [0,128)=dk, [128,512)=dv.
// ---------------------------------------------------------------------------
__global__ __launch_bounds__(256)
void tmd_edge_msg(const int* __restrict__ ei, const float* __restrict__ r_ij,
                  const float* __restrict__ d_ij, const float* __restrict__ qkv,
                  const float* __restrict__ dkv, const float* __restrict__ vecN,
                  float* __restrict__ x_agg, float* __restrict__ vec_agg,
                  int e0, int nE, int Etot) {
  int gid = blockIdx.x * blockDim.x + threadIdx.x;
  int le  = gid >> 5;
  if (le >= nE) return;
  int lane = gid & 31;
  int e    = e0 + le;
  int src  = ei[e];
  int dst  = ei[Etot + e];

  const float* qd  = qkv + (size_t)dst * 640;
  const float* kd  = qkv + (size_t)src * 640 + 128;
  const float* vd  = qkv + (size_t)src * 640 + 256;
  const float* dkp = dkv + (size_t)le * 512;
  const float* dvp = dkp + 128;
  int c = lane * 4;

  // attention logit for this head (16 channels = 4 lanes)
  float4 qv  = *(const float4*)(qd + c);
  float4 kv  = *(const float4*)(kd + c);
  float4 dk4 = *(const float4*)(dkp + c);
  float part = qv.x * kv.x * dk4.x + qv.y * kv.y * dk4.y +
               qv.z * kv.z * dk4.z + qv.w * kv.w * dk4.w;
  part += __shfl_xor(part, 1, 32);
  part += __shfl_xor(part, 2, 32);

  float r   = r_ij[e];
  float cut = (r < 5.0f) ? 0.5f * (__cosf(r * 0.6283185307179586f) + 1.0f) : 0.0f;
  float attn = part / (1.0f + __expf(-part)) * cut;   // silu(attn)*cut

  float4 vv0 = *(const float4*)(vd + c);
  float4 vv1 = *(const float4*)(vd + 128 + c);
  float4 vv2 = *(const float4*)(vd + 256 + c);
  float4 dv0 = *(const float4*)(dvp + c);
  float4 dv1 = *(const float4*)(dvp + 128 + c);
  float4 dv2 = *(const float4*)(dvp + 256 + c);

  float v1x = vv1.x * dv1.x, v1y = vv1.y * dv1.y, v1z = vv1.z * dv1.z, v1w = vv1.w * dv1.w;
  float v2x = vv2.x * dv2.x, v2y = vv2.y * dv2.y, v2z = vv2.z * dv2.z, v2w = vv2.w * dv2.w;

  float* xa = x_agg + (size_t)dst * HC + c;
  tmd_afa(&xa[0], vv0.x * dv0.x * attn);
  tmd_afa(&xa[1], vv0.y * dv0.y * attn);
  tmd_afa(&xa[2], vv0.z * dv0.z * attn);
  tmd_afa(&xa[3], vv0.w * dv0.w * attn);

  float de0 = d_ij[(size_t)e * 3 + 0];
  float de1 = d_ij[(size_t)e * 3 + 1];
  float de2 = d_ij[(size_t)e * 3 + 2];
#pragma unroll
  for (int s3 = 0; s3 < 3; ++s3) {
    float ds = (s3 == 0) ? de0 : ((s3 == 1) ? de1 : de2);
    float4 vj = *(const float4*)(vecN + ((size_t)src * 3 + s3) * HC + c);
    float* va = vec_agg + ((size_t)dst * 3 + s3) * HC + c;
    tmd_afa(&va[0], vj.x * v1x + v2x * ds);
    tmd_afa(&va[1], vj.y * v1y + v2y * ds);
    tmd_afa(&va[2], vj.z * v1z + v2z * ds);
    tmd_afa(&va[3], vj.w * v1w + v2w * ds);
  }
}

// ---------------------------------------------------------------------------
// vec_dot[n][c] = sum_s vp[n][s][c] * vp[n][s][128+c]   (vp: [N,3,384])
// ---------------------------------------------------------------------------
__global__ void tmd_vecdot(const float* __restrict__ vp, float* __restrict__ vd, int n_nodes) {
  int t = blockIdx.x * blockDim.x + threadIdx.x;
  if (t >= n_nodes * HC) return;
  int n = t >> 7, c = t & 127;
  const float* base = vp + (size_t)n * 3 * 384;
  float s = 0.0f;
#pragma unroll
  for (int s3 = 0; s3 < 3; ++s3) s += base[s3 * 384 + c] * base[s3 * 384 + 128 + c];
  vd[t] = s;
}

// ---------------------------------------------------------------------------
// dx = vec_dot*o2 + o3 ; dvec = vec3*o1 + vec_agg
// out = [dx (N*128) | dvec (N*3*128)]
// ---------------------------------------------------------------------------
__global__ void tmd_final(const float* __restrict__ o, const float* __restrict__ vdot,
                          const float* __restrict__ vp, const float* __restrict__ vec_agg,
                          float* __restrict__ out, int n_nodes) {
  int t = blockIdx.x * blockDim.x + threadIdx.x;
  if (t >= n_nodes * HC) return;
  int n = t >> 7, c = t & 127;
  const float* on = o + (size_t)n * 384;
  float o1 = on[c], o2 = on[128 + c], o3 = on[256 + c];
  out[t] = vdot[t] * o2 + o3;
  float* dvec = out + (size_t)n_nodes * HC;
#pragma unroll
  for (int s3 = 0; s3 < 3; ++s3) {
    size_t idx = ((size_t)n * 3 + s3) * HC + c;
    float v3 = vp[((size_t)n * 3 + s3) * 384 + 256 + c];
    dvec[idx] = v3 * o1 + vec_agg[idx];
  }
}

// ---------------------------------------------------------------------------
// Host-side orchestration
// ---------------------------------------------------------------------------
extern "C" void kernel_launch(void* const* d_in, const int* in_sizes, int n_in,
                              void* d_out, int out_size, void* d_ws, size_t ws_size,
                              hipStream_t stream) {
  (void)in_sizes; (void)n_in; (void)out_size; (void)ws_size;
  const float* x    = (const float*)d_in[0];
  const float* vec  = (const float*)d_in[1];
  const int*   ei   = (const int*)d_in[2];
  const float* r_ij = (const float*)d_in[3];
  const float* f_ij = (const float*)d_in[4];
  const float* d_ij = (const float*)d_in[5];
  const float* ln_g = (const float*)d_in[6];
  const float* ln_b = (const float*)d_in[7];
  const float* Wq   = (const float*)d_in[8];
  const float* bq   = (const float*)d_in[9];
  const float* Wk   = (const float*)d_in[10];
  const float* bk   = (const float*)d_in[11];
  const float* Wv   = (const float*)d_in[12];
  const float* bv   = (const float*)d_in[13];
  const float* Wo   = (const float*)d_in[14];
  const float* bo   = (const float*)d_in[15];
  const float* Wvec = (const float*)d_in[16];
  const float* Wdk  = (const float*)d_in[17];
  const float* bdk  = (const float*)d_in[18];
  const float* Wdv  = (const float*)d_in[19];
  const float* bdv  = (const float*)d_in[20];
  float* out = (float*)d_out;

  // ---- carve workspace (256B aligned slabs) ----
  char* p = (char*)d_ws;
  auto carve = [&](size_t bytes) -> void* {
    void* r = (void*)p;
    p += (bytes + 255) & ~(size_t)255;
    return r;
  };
  __bf16* xnbf   = (__bf16*)carve((size_t)N_NODES * HC * 2);
  __bf16* vecbf  = (__bf16*)carve((size_t)N_NODES * 3 * HC * 2);
  __bf16* fbf    = (__bf16*)carve((size_t)N_EDGES * RBFDIM * 2);
  __bf16* Wqkvbf = (__bf16*)carve((size_t)640 * 128 * 2);
  __bf16* Wvecbf = (__bf16*)carve((size_t)384 * 128 * 2);
  __bf16* Wdbf   = (__bf16*)carve((size_t)512 * 64 * 2);
  __bf16* Wobf   = (__bf16*)carve((size_t)384 * 128 * 2);
  float*  bqkv   = (float*)carve(640 * 4);
  float*  bd     = (float*)carve(512 * 4);
  float*  qkv    = (float*)carve((size_t)N_NODES * 640 * 4);
  float*  vp     = (float*)carve((size_t)N_NODES * 3 * 384 * 4);
  float*  x_agg  = (float*)carve((size_t)N_NODES * HC * 4);
  float*  vec_agg= (float*)carve((size_t)N_NODES * 3 * HC * 4);
  float*  dkvc   = (float*)carve((size_t)ECHUNK * 512 * 4);
  float*  vdot   = (float*)carve((size_t)N_NODES * HC * 4);
  __bf16* xaggbf = (__bf16*)carve((size_t)N_NODES * HC * 2);
  float*  obuf   = (float*)carve((size_t)N_NODES * 384 * 4);

  auto blocks = [](size_t n, int bs) { return (unsigned)((n + bs - 1) / bs); };

  // ---- pack weights (f32 -> bf16 in ws) + concat biases ----
  tmd_cvt_bf16<<<blocks(128 * 128, 256), 256, 0, stream>>>(Wq, Wqkvbf, (size_t)128 * 128);
  tmd_cvt_bf16<<<blocks(128 * 128, 256), 256, 0, stream>>>(Wk, Wqkvbf + 128 * 128, (size_t)128 * 128);
  tmd_cvt_bf16<<<blocks(384 * 128, 256), 256, 0, stream>>>(Wv, Wqkvbf + 256 * 128, (size_t)384 * 128);
  tmd_cvt_bf16<<<blocks(384 * 128, 256), 256, 0, stream>>>(Wvec, Wvecbf, (size_t)384 * 128);
  tmd_cvt_bf16<<<blocks(128 * 64, 256), 256, 0, stream>>>(Wdk, Wdbf, (size_t)128 * 64);
  tmd_cvt_bf16<<<blocks(384 * 64, 256), 256, 0, stream>>>(Wdv, Wdbf + 128 * 64, (size_t)384 * 64);
  tmd_cvt_bf16<<<blocks(384 * 128, 256), 256, 0, stream>>>(Wo, Wobf, (size_t)384 * 128);
  hipMemcpyAsync(bqkv,        bq,  128 * 4, hipMemcpyDeviceToDevice, stream);
  hipMemcpyAsync(bqkv + 128,  bk,  128 * 4, hipMemcpyDeviceToDevice, stream);
  hipMemcpyAsync(bqkv + 256,  bv,  384 * 4, hipMemcpyDeviceToDevice, stream);
  hipMemcpyAsync(bd,          bdk, 128 * 4, hipMemcpyDeviceToDevice, stream);
  hipMemcpyAsync(bd + 128,    bdv, 384 * 4, hipMemcpyDeviceToDevice, stream);

  // ---- node preprocessing ----
  tmd_ln_to_bf16<<<blocks((size_t)N_NODES * 32, 256), 256, 0, stream>>>(x, ln_g, ln_b, xnbf, N_NODES);
  tmd_cvt_bf16<<<blocks((size_t)N_NODES * 3 * HC, 256), 256, 0, stream>>>(vec, vecbf, (size_t)N_NODES * 3 * HC);
  tmd_cvt_bf16<<<blocks((size_t)N_EDGES * RBFDIM, 256), 256, 0, stream>>>(f_ij, fbf, (size_t)N_EDGES * RBFDIM);

  // ---- WMMA GEMMs: qkv = xn @ [Wq;Wk;Wv]^T + b ; vp = vec @ Wvec^T ----
  {
    dim3 g(640 / 16, N_NODES / 128);
    tmd_gemm_bf16_nt<<<g, 256, 0, stream>>>(xnbf, Wqkvbf, bqkv, qkv, 128, 640, 0);
  }
  {
    dim3 g(384 / 16, (3 * N_NODES) / 128);
    tmd_gemm_bf16_nt<<<g, 256, 0, stream>>>(vecbf, Wvecbf, nullptr, vp, 128, 384, 0);
  }

  // ---- zero accumulators (ws is poisoned; must re-init every call) ----
  tmd_zero_f32<<<blocks((size_t)N_NODES * HC, 256), 256, 0, stream>>>(x_agg, (size_t)N_NODES * HC);
  tmd_zero_f32<<<blocks((size_t)N_NODES * 3 * HC, 256), 256, 0, stream>>>(vec_agg, (size_t)N_NODES * 3 * HC);

  // ---- edge phase, chunked: dk/dv GEMM (SiLU) then message/scatter ----
  for (int ch = 0; ch < NCHUNK; ++ch) {
    dim3 g(512 / 16, ECHUNK / 128);
    tmd_gemm_bf16_nt<<<g, 256, 0, stream>>>(fbf + (size_t)ch * ECHUNK * RBFDIM, Wdbf, bd,
                                            dkvc, 64, 512, 1);
    tmd_edge_msg<<<blocks((size_t)ECHUNK * 32, 256), 256, 0, stream>>>(
        ei, r_ij, d_ij, qkv, dkvc, vec, x_agg, vec_agg, ch * ECHUNK, ECHUNK, N_EDGES);
  }

  // ---- output projection + epilogue ----
  tmd_vecdot<<<blocks((size_t)N_NODES * HC, 256), 256, 0, stream>>>(vp, vdot, N_NODES);
  tmd_cvt_bf16<<<blocks((size_t)N_NODES * HC, 256), 256, 0, stream>>>(x_agg, xaggbf, (size_t)N_NODES * HC);
  {
    dim3 g(384 / 16, N_NODES / 128);
    tmd_gemm_bf16_nt<<<g, 256, 0, stream>>>(xaggbf, Wobf, bo, obuf, 128, 384, 0);
  }
  tmd_final<<<blocks((size_t)N_NODES * HC, 256), 256, 0, stream>>>(obuf, vdot, vp, vec_agg, out, N_NODES);
}
